// HMMLikelihoodLoss_28303834480643
// MI455X (gfx1250) — matrix operand order repaired and checked by reading, not compiled, not run on hardware
//
#include <hip/hip_runtime.h>
#include <hip/hip_bf16.h>

typedef __attribute__((ext_vector_type(16))) _Float16 v16h;
typedef __attribute__((ext_vector_type(8)))  _Float16 v8h;
typedef __attribute__((ext_vector_type(8)))  float    v8f;

constexpr int S = 64;      // states
constexpr int E = 1024;    // emissions
constexpr int B = 256;     // batch
constexpr int T = 4096;    // sequence length

constexpr int EP = E + 1;  // padded LDS emission row stride (bank-conflict-free gathers)
constexpr int AP = S + 8;  // padded LDS alpha row stride: 72 halfs = 144 B (16B aligned,
                           // n*36 mod 64 distinct for n=0..15 -> conflict-free b128 loads)

#define INV_LN2 1.4426950408889634f
#define LN2     0.6931471805599453f

// ---------------- helpers ----------------

__device__ __forceinline__ v16h mk16(v8h lo, v8h hi) {
  return __builtin_shufflevector(lo, hi, 0,1,2,3,4,5,6,7,8,9,10,11,12,13,14,15);
}

// 16-lane butterfly max via DPP ROW_XMASK (0x160 | mask): pure VALU, no LDS.
template <int CTRL>
__device__ __forceinline__ float dpp_xor_max(float x) {
  int xi = __builtin_bit_cast(int, x);
  int yi = __builtin_amdgcn_update_dpp(xi, xi, CTRL, 0xF, 0xF, true);
  return fmaxf(x, __builtin_bit_cast(float, yi));
}

__device__ __forceinline__ float halfrow_max(float m) {
  m = dpp_xor_max<0x161>(m);   // ROW_XMASK 1
  m = dpp_xor_max<0x162>(m);   // ROW_XMASK 2
  m = dpp_xor_max<0x164>(m);   // ROW_XMASK 4
  m = dpp_xor_max<0x168>(m);   // ROW_XMASK 8
  return m;
}

// ---------------- prologue: parameter normalization ----------------

// exp(log_softmax(log_trans)) stored TRANSPOSED f16: Tt[j][i]; column-contiguous
// in i so each WMMA B-fragment lane load is a contiguous 16-half run.
__global__ void hmm_norm_trans(const float* __restrict__ lt, _Float16* __restrict__ Tt) {
  __shared__ float red[S];
  const int i = blockIdx.x, j = threadIdx.x;
  const float x = lt[i * S + j];
  red[j] = x; __syncthreads();
  for (int off = 32; off > 0; off >>= 1) { if (j < off) red[j] = fmaxf(red[j], red[j + off]); __syncthreads(); }
  const float mx = red[0]; __syncthreads();
  red[j] = __expf(x - mx); __syncthreads();
  for (int off = 32; off > 0; off >>= 1) { if (j < off) red[j] += red[j + off]; __syncthreads(); }
  const float lse = mx + __logf(red[0]);
  Tt[j * S + i] = (_Float16)__expf(x - lse);
}

// log2-domain normalized initial distribution.
__global__ void hmm_norm_init(const float* __restrict__ li, float* __restrict__ In) {
  __shared__ float red[S];
  const int j = threadIdx.x;
  const float x = li[j];
  red[j] = x; __syncthreads();
  for (int off = 32; off > 0; off >>= 1) { if (j < off) red[j] = fmaxf(red[j], red[j + off]); __syncthreads(); }
  const float mx = red[0]; __syncthreads();
  red[j] = __expf(x - mx); __syncthreads();
  for (int off = 32; off > 0; off >>= 1) { if (j < off) red[j] += red[j + off]; __syncthreads(); }
  In[j] = (x - (mx + __logf(red[0]))) * INV_LN2;
}

// log2-domain normalized emissions, f32 [S][E].
__global__ void hmm_norm_emit(const float* __restrict__ le, float* __restrict__ En) {
  __shared__ float red[256];
  const int s = blockIdx.x, t = threadIdx.x;
  float mx = -INFINITY;
  for (int e = t; e < E; e += 256) mx = fmaxf(mx, le[s * E + e]);
  red[t] = mx; __syncthreads();
  for (int off = 128; off > 0; off >>= 1) { if (t < off) red[t] = fmaxf(red[t], red[t + off]); __syncthreads(); }
  mx = red[0]; __syncthreads();
  float sm = 0.0f;
  for (int e = t; e < E; e += 256) sm += __expf(le[s * E + e] - mx);
  red[t] = sm; __syncthreads();
  for (int off = 128; off > 0; off >>= 1) { if (t < off) red[t] += red[t + off]; __syncthreads(); }
  const float lse = mx + __logf(red[0]);
  for (int e = t; e < E; e += 256) En[s * E + e] = (le[s * E + e] - lse) * INV_LN2;
}

// obs [B][T] -> obsT [T][B]: per-step observation fetch becomes two int4 broadcasts.
__global__ void hmm_transpose_obs(const int* __restrict__ obs, int* __restrict__ obsT) {
  __shared__ int tile[32][33];
  const int tBase = blockIdx.x * 32, bBase = blockIdx.y * 32;
  const int tx = threadIdx.x, ty = threadIdx.y;
  tile[ty][tx] = obs[(bBase + ty) * T + tBase + tx];
  __syncthreads();
  obsT[(tBase + ty) * B + bBase + tx] = tile[tx][ty];
}

// ---------------- main forward-recursion kernel ----------------
// 16 workgroups x 32 threads: ONE wave owns 16 batch rows end-to-end.
// Scaled linear alpha a[r][i] = exp2(log2_alpha - m2[r]) in f16 LDS (padded);
// per-row log2-scale m2 in registers (replicated across each 16-lane half).
// Per step: 8x v_wmma_f32_16x16x32_f16 (4 j-tiles x K=64) + native log2/exp2
// epilogue + DPP butterfly row-max. Zero barriers in the T-loop (same-wave
// LDS ops are in-order). 256KB emission table cached in LDS, bank-padded.
__global__ __launch_bounds__(32)
void hmm_forward_kernel(const int* __restrict__ obsT, const float* __restrict__ En,
                        const float* __restrict__ In, const _Float16* __restrict__ Tt,
                        float* __restrict__ partials) {
  __shared__ float    sEn[S * EP];    // 64*1025*4 = 256.25 KB
  __shared__ _Float16 sA[16 * AP];    // 16*72*2   = 2.25 KB
  __shared__ float    sMfin[16];
  __shared__ float    sLP[16];

  const int lane   = threadIdx.x & 31;
  const int h      = lane >> 4;     // half of wave
  const int n      = lane & 15;     // column within tile / A-frag row
  const int batch0 = blockIdx.x * 16;

  // ---- cache normalized emissions into padded LDS ----
  for (int idx = lane * 4; idx < S * E; idx += 128) {
    const float4 f = *(const float4*)(En + idx);
    float* d = sEn + (idx >> 10) * EP + (idx & 1023);
    d[0] = f.x; d[1] = f.y; d[2] = f.z; d[3] = f.w;
  }

  // ---- time-invariant B fragments: lane = column N = jt*16+n,
  //      K = kc*32 + h*16 + 0..15 contiguous in Tt column ----
  v16h bfrag[4][2];
  for (int jt = 0; jt < 4; ++jt) {
    const _Float16* col = Tt + (jt * 16 + n) * S;
    for (int kc = 0; kc < 2; ++kc) {
      const v8h* p = (const v8h*)(col + kc * 32 + h * 16);
      bfrag[jt][kc] = mk16(p[0], p[1]);
    }
  }

  float m2[8];                       // per-row running log2-scale (rows 8h..8h+7)
  for (int v = 0; v < 8; ++v) m2[v] = 0.0f;

  // ---- t = 0: log2_alpha0 = log2_init + emit2 ----
  {
    const int4 oA = *(const int4*)(obsT + batch0 + 8 * h);
    const int4 oB = *(const int4*)(obsT + batch0 + 8 * h + 4);
    const int o[8] = {oA.x, oA.y, oA.z, oA.w, oB.x, oB.y, oB.z, oB.w};
    float z[4][8];
    for (int jt = 0; jt < 4; ++jt) {
      const float inz = In[jt * 16 + n];
      const float* colbase = sEn + (jt * 16 + n) * EP;
      for (int v = 0; v < 8; ++v) z[jt][v] = inz + colbase[o[v]];
    }
    for (int v = 0; v < 8; ++v) {
      float c = fmaxf(fmaxf(z[0][v], z[1][v]), fmaxf(z[2][v], z[3][v]));
      c = halfrow_max(c);
      m2[v] += c;
      const int r = 8 * h + v;
      for (int jt = 0; jt < 4; ++jt)
        sA[r * AP + jt * 16 + n] = (_Float16)__builtin_amdgcn_exp2f(z[jt][v] - c);
    }
  }

  // ---- forward recursion (no barriers: single wave, in-order LDS) ----
  for (int t = 1; t < T; ++t) {
    const int* op = obsT + t * B + batch0 + 8 * h;
    const int4 oA = *(const int4*)(op);
    const int4 oB = *(const int4*)(op + 4);
    const int o[8] = {oA.x, oA.y, oA.z, oA.w, oB.x, oB.y, oB.z, oB.w};

    // A fragments: lane row M=n; elems 0-15: K=kc*32+h*16+0..15 (contiguous, matches B).
    const _Float16* row = sA + n * AP;
    const v16h a0 = mk16(*(const v8h*)(row + h * 16),      *(const v8h*)(row + 8 + h * 16));
    const v16h a1 = mk16(*(const v8h*)(row + 32 + h * 16), *(const v8h*)(row + 40 + h * 16));

    // y = a_lin @ Tnorm, then z2 = log2(y) + emit2 (native v_log_f32).
    float z[4][8];
    for (int jt = 0; jt < 4; ++jt) {
      v8f acc = {};
      acc = __builtin_amdgcn_wmma_f32_16x16x32_f16(false, a0, false, bfrag[jt][0],
                                                   (short)0, acc, false, false);
      acc = __builtin_amdgcn_wmma_f32_16x16x32_f16(false, a1, false, bfrag[jt][1],
                                                   (short)0, acc, false, false);
      const float* colbase = sEn + (jt * 16 + n) * EP;
      for (int v = 0; v < 8; ++v)
        z[jt][v] = __builtin_amdgcn_logf(acc[v]) + colbase[o[v]];
    }

    // exact per-row rescale: row max over 64 columns (4 tiles + DPP butterfly).
    for (int v = 0; v < 8; ++v) {
      float c = fmaxf(fmaxf(z[0][v], z[1][v]), fmaxf(z[2][v], z[3][v]));
      c = halfrow_max(c);
      m2[v] += c;
      const int r = 8 * h + v;
      for (int jt = 0; jt < 4; ++jt)
        sA[r * AP + jt * 16 + n] = (_Float16)__builtin_amdgcn_exp2f(z[jt][v] - c);
    }
  }

  // ---- finalize: log p(x_b) = ln2 * (m2_b + log2(sum_j a[b][j])) ----
  if (n == 0)
    for (int v = 0; v < 8; ++v) sMfin[8 * h + v] = m2[v];
  __syncthreads();
  if (lane < 16) {
    float s = 0.0f;
    for (int k = 0; k < S; ++k) s += (float)sA[lane * AP + k];
    sLP[lane] = (sMfin[lane] + __builtin_amdgcn_logf(s)) * LN2;
  }
  __syncthreads();
  if (lane == 0) {
    float s = 0.0f;
    for (int r = 0; r < 16; ++r) s += sLP[r];
    partials[blockIdx.x] = s;
  }
}

__global__ void hmm_final_reduce(const float* __restrict__ partials, float* __restrict__ out) {
  if (threadIdx.x == 0) {
    float s = 0.0f;
    for (int i = 0; i < 16; ++i) s += partials[i];
    out[0] = -s / (float)B;
  }
}

// ---------------- launch ----------------

extern "C" void kernel_launch(void* const* d_in, const int* in_sizes, int n_in,
                              void* d_out, int out_size, void* d_ws, size_t ws_size,
                              hipStream_t stream) {
  const int*   obs = (const int*)d_in[0];    // [B][T]
  const float* li  = (const float*)d_in[1];  // [S]
  const float* lt  = (const float*)d_in[2];  // [S][S]
  const float* le  = (const float*)d_in[3];  // [S][E]
  float* out = (float*)d_out;

  char* ws = (char*)d_ws;
  _Float16* Tt       = (_Float16*)(ws);                      //   8 KB f16 [S][S] transposed
  float*    In       = (float*)(ws + 8192);                  // 256 B (log2 domain)
  float*    En       = (float*)(ws + 16384);                 // 256 KB f32 [S][E] (log2 domain)
  int*      obsT     = (int*)(ws + 16384 + 262144);          //   4 MB int [T][B]
  float*    partials = (float*)(ws + 16384 + 262144 + 4194304);

  hmm_norm_trans<<<S, S, 0, stream>>>(lt, Tt);
  hmm_norm_init <<<1, S, 0, stream>>>(li, In);
  hmm_norm_emit <<<S, 256, 0, stream>>>(le, En);
  hmm_transpose_obs<<<dim3(T / 32, B / 32), dim3(32, 32), 0, stream>>>(obs, obsT);
  hmm_forward_kernel<<<B / 16, 32, 0, stream>>>(obsT, En, In, Tt, partials);
  hmm_final_reduce<<<1, 32, 0, stream>>>(partials, out);
}
